// ApproxMiniS4D_69758858822347
// MI455X (gfx1250) — matrix operand order repaired and checked by reading, not compiled, not run on hardware
//
#include <hip/hip_runtime.h>
#include <hip/hip_bf16.h>
#include <hip/hip_fp16.h>

typedef __attribute__((ext_vector_type(16))) _Float16 v16h;
typedef __attribute__((ext_vector_type(8)))  _Float16 v8h;
typedef __attribute__((ext_vector_type(8)))  float    v8f;

#define D_MODEL 128
#define D_STATE 64
#define LSEQ    1024
#define BATCH   32
#define NCOL    (BATCH * LSEQ)   // 32768
#define KPAD    32
#define KROW    (LSEQ + KPAD)    // 1056 halfs per padded K row

__device__ __forceinline__ v16h join16(v8h lo, v8h hi) {
  return __builtin_shufflevector(lo, hi, 0, 1, 2, 3, 4, 5, 6, 7,
                                         8, 9, 10, 11, 12, 13, 14, 15);
}

// ---------------------------------------------------------------------------
// K0: per (d,n) SSM parameters:  w = Cc * B_bar,  (a,b) = A*dt (log of A_bar)
// ---------------------------------------------------------------------------
__global__ void k_params(const float* __restrict__ log_dt,
                         const float* __restrict__ log_A_real,
                         const float* __restrict__ A_imag,
                         const float* __restrict__ B_re, const float* __restrict__ B_im,
                         const float* __restrict__ C_re, const float* __restrict__ C_im,
                         float* __restrict__ w_re, float* __restrict__ w_im,
                         float* __restrict__ aa,   float* __restrict__ bb) {
  int i = blockIdx.x * blockDim.x + threadIdx.x;
  if (i >= D_MODEL * D_STATE) return;
  int d = i / D_STATE;
  float dt = __expf(log_dt[d]);
  float ar = -__expf(log_A_real[i]);
  float ai = A_imag[i];
  float a = ar * dt, b = ai * dt;           // log(A_bar)
  float ea = __expf(a);
  float sb, cb;
  __sincosf(b, &sb, &cb);
  float Abr = ea * cb, Abi = ea * sb;       // A_bar
  float nr = Abr - 1.0f, ni = Abi;          // (A_bar - 1) / A
  float den = ar * ar + ai * ai;
  float qr = (nr * ar + ni * ai) / den;
  float qi = (ni * ar - nr * ai) / den;
  float Br = B_re[i], Bi = B_im[i];         // * Bc
  float Bbr = qr * Br - qi * Bi;
  float Bbi = qr * Bi + qi * Br;
  float Cr = C_re[i], Ci = C_im[i];         // w = Cc * B_bar
  w_re[i] = Cr * Bbr - Ci * Bbi;
  w_im[i] = Cr * Bbi + Ci * Bbr;
  aa[i] = a;
  bb[i] = b;
}

// ---------------------------------------------------------------------------
// K1: padded K rows:  Khp[d][0..31] = 0,  Khp[d][32+l] = K[d,l]  (f16)
// ---------------------------------------------------------------------------
__global__ void k_kernelK(const float* __restrict__ w_re, const float* __restrict__ w_im,
                          const float* __restrict__ aa,   const float* __restrict__ bb,
                          _Float16* __restrict__ Khp) {
  int i = blockIdx.x * blockDim.x + threadIdx.x;      // d*KROW + p
  if (i >= D_MODEL * KROW) return;
  int d = i / KROW;
  int p = i - d * KROW;
  if (p < KPAD) { Khp[i] = (_Float16)0.0f; return; }
  float lf = (float)(p - KPAD);
  int base = d * D_STATE;
  float acc = 0.0f;
  #pragma unroll 4
  for (int n = 0; n < D_STATE; ++n) {
    float e = __expf(aa[base + n] * lf);
    float s, c;
    sincosf(bb[base + n] * lf, &s, &c);
    acc += e * (w_re[base + n] * c - w_im[base + n] * s);
  }
  Khp[i] = (_Float16)acc;
}

// ---------------------------------------------------------------------------
// K2a: u (B,d,L) f32  ->  Ut[d][b][l] f16  (conv B-matrix: K=l contiguous/lane)
// ---------------------------------------------------------------------------
__global__ void k_u2h(const float* __restrict__ u, _Float16* __restrict__ Ut) {
  int i = blockIdx.x * blockDim.x + threadIdx.x;      // (d*BATCH + b)*LSEQ + l
  if (i >= D_MODEL * LSEQ * BATCH) return;
  int l = i & (LSEQ - 1);
  int b = (i >> 10) & (BATCH - 1);
  int d = i >> 15;
  Ut[i] = (_Float16)u[((size_t)b * D_MODEL + d) * LSEQ + l];
}

// K2b: W1 f32 -> f16
__global__ void k_w2h(const float* __restrict__ W1, _Float16* __restrict__ Wh) {
  int i = blockIdx.x * blockDim.x + threadIdx.x;
  if (i < 2 * D_MODEL * D_MODEL) Wh[i] = (_Float16)W1[i];
}

// ---------------------------------------------------------------------------
// K2c: precompute Toeplitz A-tiles in WMMA fragment order.
// Tile(d, diff):  A[m,k] = K[d, diff + m - k]  (zeros via padded K row).
// Per lane (ISA 16-bit A 16x32 layout): m = lane&15, k(e) = e + (e&8?8:0) + hi*8.
// Stored 16 contiguous halfs per lane -> k_conv loads 2x b128.
// blockIdx.x = diffIdx (0..63, diff = 16*diffIdx), blockIdx.y = d. 32 threads.
// ---------------------------------------------------------------------------
__global__ void __launch_bounds__(32)
k_atiles(const _Float16* __restrict__ Khp, _Float16* __restrict__ Atiles) {
  const int diffIdx = blockIdx.x;
  const int d       = blockIdx.y;
  const int lane    = threadIdx.x & 31;
  const int hi      = lane >> 4;
  const int mn      = lane & 15;
  const int diff    = diffIdx * 16;
  const _Float16* row = Khp + (size_t)d * KROW + KPAD;   // row[idx], idx >= -32 ok
  v16h t;
  #pragma unroll
  for (int e = 0; e < 16; ++e) {
    int k = e + ((e & 8) ? 8 : 0) + hi * 8;
    t[e] = row[diff + mn - k];
  }
  *(v16h*)(Atiles + (((size_t)d * 64 + diffIdx) * 32 + lane) * 16) = t;
}

// ---------------------------------------------------------------------------
// Chebyshev poly6 GELU on [-5, 15]
// ---------------------------------------------------------------------------
__device__ __forceinline__ float cheb_gelu(float x) {
  float t = (x - 5.0f) * 0.1f;
  float t0 = 1.0f, t1 = t, t2;
  float r = fmaf(8.094460228971f, t, 6.04831644882f);
  t2 = fmaf(2.0f * t, t1, -t0); r = fmaf( 1.371512430522f, t2, r); t0 = t1; t1 = t2;
  t2 = fmaf(2.0f * t, t1, -t0); r = fmaf(-0.740775295685f, t2, r); t0 = t1; t1 = t2;
  t2 = fmaf(2.0f * t, t1, -t0); r = fmaf( 0.134773988002f, t2, r); t0 = t1; t1 = t2;
  t2 = fmaf(2.0f * t, t1, -t0); r = fmaf( 0.126019270103f, t2, r); t0 = t1; t1 = t2;
  t2 = fmaf(2.0f * t, t1, -t0); r = fmaf(-0.239569101196f, t2, r);
  return r;
}

// ---------------------------------------------------------------------------
// K3: depthwise causal conv as triangular-Toeplitz GEMM via WMMA,
//     fused with skip (+D*u) and Chebyshev GELU  ->  X2[col][d] f16
// Block: 64 threads (2 waves). wave0 -> batch cols 0-15, wave1 -> 16-31.
// blockIdx.x = l-tile (0..63), blockIdx.y = d (0..127).
// Inner loop: 4x global_load_b128 + 1x v_wmma, branch-free.
// ---------------------------------------------------------------------------
__global__ void __launch_bounds__(64)
k_conv(const _Float16* __restrict__ Atiles, const _Float16* __restrict__ Ut,
       const float* __restrict__ u, const float* __restrict__ Dp,
       _Float16* __restrict__ X2) {
  const int mtile = blockIdx.x;
  const int d     = blockIdx.y;
  const int lane  = threadIdx.x & 31;
  const int wave  = threadIdx.x >> 5;
  const int hi    = lane >> 4;
  const int mn    = lane & 15;

  const int L0   = mtile * 16;
  const int bcol = wave * 16 + mn;                 // batch column this lane owns
  const _Float16* Ub = Ut + ((size_t)d * BATCH + bcol) * LSEQ + hi * 8;
  const _Float16* Ab = Atiles + ((size_t)d * 64 * 32 + lane) * 16;

  v8f c = {};
  int diffIdx = mtile;                             // diff = L0 - j0, step -32
  for (int j0 = 0; j0 < L0 + 16; j0 += 32, diffIdx -= 2) {
    const _Float16* ap = Ab + (size_t)diffIdx * 32 * 16;
    v16h af = join16(*(const v8h*)ap, *(const v8h*)(ap + 8));
    const _Float16* bp = Ub + j0;
    v16h bf = join16(*(const v8h*)bp, *(const v8h*)(bp + 16));
    __builtin_prefetch(bp + 32, 0, 1);             // next chunk (speculative ok)
    c = __builtin_amdgcn_wmma_f32_16x16x32_f16(false, af, false, bf,
                                               (short)0, c, false, false);
  }

  // Epilogue: skip + GELU -> X2[col][d]  (C: lanes<16 rows 0-7, >=16 rows 8-15)
  const float Dd = Dp[d];
  const float* ucol = u + ((size_t)bcol * D_MODEL + d) * LSEQ;
  _Float16* xb = X2 + ((size_t)bcol * LSEQ + L0 + hi * 8) * D_MODEL + d;
  #pragma unroll
  for (int v = 0; v < 8; ++v) {
    float skip = c[v] + Dd * ucol[L0 + hi * 8 + v];
    xb[(size_t)v * D_MODEL] = (_Float16)cheb_gelu(skip);
  }
}

// ---------------------------------------------------------------------------
// K4: 1x1 conv GEMM via WMMA + bias + linear gate + mean-pool (atomicAdd)
// Block: 32 threads. blockIdx.x = col tile (0..2047), blockIdx.y = ot (0..7).
// Rows ot*16+m are 'a' channels, +128 the gate channels; shared B-fragment.
// ---------------------------------------------------------------------------
__global__ void __launch_bounds__(32)
k_gemm2(const _Float16* __restrict__ Wh, const _Float16* __restrict__ X2,
        const float* __restrict__ b1, float* __restrict__ pooled) {
  const int ntile = blockIdx.x;
  const int ot    = blockIdx.y;
  const int lane  = threadIdx.x & 31;
  const int hi    = lane >> 4;
  const int mn    = lane & 15;
  const int colbase = ntile * 16;
  const int batch   = colbase >> 10;     // 64 col-tiles per batch, never straddles

  const _Float16* wa0 = Wh + ((size_t)(ot * 16 + mn)) * D_MODEL + hi * 8;
  const _Float16* wb0 = wa0 + (size_t)128 * D_MODEL;
  const _Float16* xp0 = X2 + ((size_t)(colbase + mn)) * D_MODEL + hi * 8;

  v8f ca = {}, cg = {};
  #pragma unroll
  for (int kc = 0; kc < D_MODEL; kc += 32) {
    v16h aA = join16(*(const v8h*)(wa0 + kc), *(const v8h*)(wa0 + kc + 16));
    v16h aB = join16(*(const v8h*)(wb0 + kc), *(const v8h*)(wb0 + kc + 16));
    v16h bf = join16(*(const v8h*)(xp0 + kc), *(const v8h*)(xp0 + kc + 16));
    ca = __builtin_amdgcn_wmma_f32_16x16x32_f16(false, aA, false, bf,
                                                (short)0, ca, false, false);
    cg = __builtin_amdgcn_wmma_f32_16x16x32_f16(false, aB, false, bf,
                                                (short)0, cg, false, false);
  }

  #pragma unroll
  for (int v = 0; v < 8; ++v) {
    int chan = ot * 16 + v + hi * 8;
    float aval = ca[v] + b1[chan];
    float bval = cg[v] + b1[128 + chan];
    float gate = fminf(fmaxf(fmaf(0.25f, bval, 0.5f), 0.0f), 1.0f);
    float g = aval * gate;
    g += __shfl_xor(g, 1, 32);
    g += __shfl_xor(g, 2, 32);
    g += __shfl_xor(g, 4, 32);
    g += __shfl_xor(g, 8, 32);
    if (mn == 0)
      atomicAdd(&pooled[batch * D_MODEL + chan], g * (1.0f / (float)LSEQ));
  }
}

// ---------------------------------------------------------------------------
// K5: out[b] = pooled[b,:] . Wdec + bdec
// ---------------------------------------------------------------------------
__global__ void k_final(const float* __restrict__ pooled, const float* __restrict__ Wdec,
                        const float* __restrict__ bdec, float* __restrict__ out) {
  int b = threadIdx.x;
  if (b >= BATCH) return;
  float s = 0.0f;
  #pragma unroll 4
  for (int c = 0; c < D_MODEL; ++c) s += pooled[b * D_MODEL + c] * Wdec[c];
  out[b] = s + bdec[0];
}

// ---------------------------------------------------------------------------
extern "C" void kernel_launch(void* const* d_in, const int* in_sizes, int n_in,
                              void* d_out, int out_size, void* d_ws, size_t ws_size,
                              hipStream_t stream) {
  (void)in_sizes; (void)n_in; (void)out_size; (void)ws_size;
  const float* u      = (const float*)d_in[0];
  const float* log_dt = (const float*)d_in[1];
  const float* log_A  = (const float*)d_in[2];
  const float* A_imag = (const float*)d_in[3];
  const float* B_re   = (const float*)d_in[4];
  const float* B_im   = (const float*)d_in[5];
  const float* C_re   = (const float*)d_in[6];
  const float* C_im   = (const float*)d_in[7];
  const float* Dp     = (const float*)d_in[8];
  const float* W1     = (const float*)d_in[9];
  const float* b1     = (const float*)d_in[10];
  const float* Wdec   = (const float*)d_in[11];
  const float* bdec   = (const float*)d_in[12];
  float* out = (float*)d_out;

  char* ws = (char*)d_ws;
  size_t off = 0;
  auto alloc = [&](size_t bytes) -> char* {
    off = (off + 255) & ~(size_t)255;
    char* p = ws + off;
    off += bytes;
    return p;
  };
  float*    w_re   = (float*)alloc((size_t)D_MODEL * D_STATE * 4);
  float*    w_im   = (float*)alloc((size_t)D_MODEL * D_STATE * 4);
  float*    aa     = (float*)alloc((size_t)D_MODEL * D_STATE * 4);
  float*    bb     = (float*)alloc((size_t)D_MODEL * D_STATE * 4);
  _Float16* Khp    = (_Float16*)alloc((size_t)D_MODEL * KROW * 2);
  _Float16* Atiles = (_Float16*)alloc((size_t)D_MODEL * 64 * 32 * 16 * 2);
  _Float16* Ut     = (_Float16*)alloc((size_t)D_MODEL * BATCH * LSEQ * 2);
  _Float16* X2     = (_Float16*)alloc((size_t)NCOL * D_MODEL * 2);
  _Float16* Wh     = (_Float16*)alloc((size_t)2 * D_MODEL * D_MODEL * 2);
  float*    pooled = (float*)alloc((size_t)BATCH * D_MODEL * 4);

  hipMemsetAsync(pooled, 0, (size_t)BATCH * D_MODEL * 4, stream);

  k_params<<<(D_MODEL * D_STATE + 255) / 256, 256, 0, stream>>>(
      log_dt, log_A, A_imag, B_re, B_im, C_re, C_im, w_re, w_im, aa, bb);

  k_kernelK<<<(D_MODEL * KROW + 255) / 256, 256, 0, stream>>>(w_re, w_im, aa, bb, Khp);

  k_u2h<<<(D_MODEL * LSEQ * BATCH) / 256, 256, 0, stream>>>(u, Ut);
  k_w2h<<<(2 * D_MODEL * D_MODEL) / 256, 256, 0, stream>>>(W1, Wh);

  k_atiles<<<dim3(64, D_MODEL), 32, 0, stream>>>(Khp, Atiles);

  k_conv<<<dim3(LSEQ / 16, D_MODEL), 64, 0, stream>>>(Atiles, Ut, u, Dp, X2);

  k_gemm2<<<dim3(NCOL / 16, 8), 32, 0, stream>>>(Wh, X2, b1, pooled);

  k_final<<<1, 32, 0, stream>>>(pooled, Wdec, bdec, out);
}
